// Attention_61022895341857
// MI455X (gfx1250) — matrix operand order repaired and checked by reading, not compiled, not run on hardware
//
#include <hip/hip_runtime.h>
#include <hip/hip_bf16.h>
#include <stdint.h>

typedef unsigned short u16;
typedef __attribute__((ext_vector_type(16))) __bf16 v16bf;
typedef __attribute__((ext_vector_type(8)))  float  v8f;

#define S_LEN 2048
#define HID   576
#define NH    9
#define NKV   3
#define HD    64
#define NB    4
#define TOK   (NB * S_LEN)          // 8192
#define PSTRIDE 40                  // LDS row strides (u16): 16B-aligned, low-conflict
#define KSTRIDE 72
#define VSTRIDE 40
#define WSTRIDE 40

union ABPack { uint4 u[2]; v16bf v; };

// A-matrix 16x32 bf16: lane row = lane%16; elem j -> K = 2*koff+j,
// koff = 8*(lane/16). Two 16B loads.
__device__ __forceinline__ v16bf load_a(const u16* rowptr, int koff) {
    ABPack x;
    x.u[0] = *(const uint4*)(rowptr + 2 * koff);
    x.u[1] = *(const uint4*)(rowptr + 2 * koff + 8);
    return x.v;
}
// B-matrix 32x16 bf16: lane col = lane%16; elem j -> K = 16*(lane/16)+j.
// Caller pre-offsets by 16*(lane/16): 16 contiguous bf16.
__device__ __forceinline__ v16bf load_b(const u16* kptr) {
    ABPack x;
    x.u[0] = *(const uint4*)(kptr);
    x.u[1] = *(const uint4*)(kptr + 8);
    return x.v;
}

__device__ __forceinline__ v8f wmma_bf16(v16bf a, v16bf b, v8f c) {
    return __builtin_amdgcn_wmma_f32_16x16x32_bf16(false, a, false, b, (short)0, c,
                                                   false, false);
}

__device__ __forceinline__ u16 f2bf(float f) {
    uint32_t u = __builtin_bit_cast(uint32_t, f);
    u += 0x7FFFu + ((u >> 16) & 1u);   // round-to-nearest-even
    return (u16)(u >> 16);
}

// ---------------------------------------------------------------------------
// Kernel 0: fp32 -> bf16 conversion
// ---------------------------------------------------------------------------
__global__ void cvt_f32_bf16(const float* __restrict__ src, u16* __restrict__ dst, int n) {
    int i = blockIdx.x * blockDim.x + threadIdx.x;
    if (i < n) dst[i] = f2bf(src[i]);
}

// ---------------------------------------------------------------------------
// Kernel 1: QKV projection + RoPE. One wave computes a 16-token x 64-dim
// (full head) tile so RoPE pairing (d <-> d±32) stays in-register.
// The 64x32 weight chunk for the current k-step is staged in LDS once per
// block and shared by all 4 waves.
// grid.y: 0..8 = Q head, 9..11 = K kv-head, 12..14 = V kv-head.
// ---------------------------------------------------------------------------
__global__ __launch_bounds__(128) void qkv_rope_kernel(
    const u16* __restrict__ Xb,  const u16* __restrict__ Wqb,
    const u16* __restrict__ Wkb, const u16* __restrict__ Wvb,
    u16* __restrict__ Qr, u16* __restrict__ Kr, u16* __restrict__ Vt)
{
    __shared__ __align__(16) u16 Wl[64 * WSTRIDE];

    const int tid  = threadIdx.x;
    const int lane = tid & 31;
    const int wave = tid >> 5;
    const int l16  = lane & 15;
    const int half = lane >> 4;
    const int koff = half * 8;
    const int t0   = (blockIdx.x * 4 + wave) * 16;
    const int grp  = blockIdx.y;

    const u16* W; int hh; int mode;
    if (grp < 9)       { W = Wqb; hh = grp;      mode = 0; }
    else if (grp < 12) { W = Wkb; hh = grp - 9;  mode = 1; }
    else               { W = Wvb; hh = grp - 12; mode = 2; }
    const int e0 = hh * HD;

    v8f acc[4] = {};
    const u16* arow = Xb + (size_t)(t0 + l16) * HID;

    for (int kb = 0; kb < HID; kb += 32) {
        // cooperative stage of W[e0..e0+63][kb..kb+31] (4KB) into LDS
#pragma unroll
        for (int i = tid; i < 256; i += 128) {
            int row = i >> 2, part = i & 3;
            const u16* src = W + (size_t)(e0 + row) * HID + kb + part * 8;
            if (kb + 32 < HID) __builtin_prefetch(src + 32, 0, 1);
            *(uint4*)(Wl + row * WSTRIDE + part * 8) = *(const uint4*)src;
        }
        __syncthreads();

        v16bf a = load_a(arow + kb, koff);
#pragma unroll
        for (int j = 0; j < 4; ++j)
            acc[j] = wmma_bf16(a, load_b(Wl + (j * 16 + l16) * WSTRIDE + half * 16),
                               acc[j]);
        __syncthreads();
    }

    const int b     = t0 >> 11;       // t0 / 2048
    const int sbase = t0 & (S_LEN - 1);

    if (mode == 2) {                  // V: store transposed [d][s]
        u16* base = Vt + (size_t)(b * NKV + hh) * HD * S_LEN;
#pragma unroll
        for (int j = 0; j < 4; ++j) {
            int d = j * 16 + l16;
#pragma unroll
            for (int r = 0; r < 8; ++r)
                base[(size_t)d * S_LEN + sbase + r + 8 * half] = f2bf(acc[j][r]);
        }
        return;
    }

    // RoPE: d<32: x*cos - x[d+32]*sin ; d>=32: x*cos + x[d-32]*sin
    v8f rot[4];
#pragma unroll
    for (int j = 0; j < 4; ++j) {
        int d = j * 16 + l16;
        float invf = __expf((float)(d & 31) * -0.28782313662425572f); // -ln(1e4)/32
#pragma unroll
        for (int r = 0; r < 8; ++r) {
            int s = sbase + r + 8 * half;
            float sn, cs;
            __sincosf((float)s * invf, &sn, &cs);
            float v  = acc[j][r];
            float pv = acc[(j + 2) & 3][r];
            rot[j][r] = (j < 2) ? (v * cs - pv * sn) : (v * cs + pv * sn);
        }
    }
    u16* base = (mode == 0) ? (Qr + (size_t)(b * NH + hh) * S_LEN * HD)
                            : (Kr + (size_t)(b * NKV + hh) * S_LEN * HD);
#pragma unroll
    for (int j = 0; j < 4; ++j)
#pragma unroll
        for (int r = 0; r < 8; ++r)
            base[(size_t)(sbase + r + 8 * half) * HD + j * 16 + l16] = f2bf(rot[j][r]);
}

// ---------------------------------------------------------------------------
// Kernel 2: flash attention. 8 waves per block share one (b,h); each wave owns
// a 16-row Q tile. K/V stream in 32-key chunks staged ONCE per block in LDS
// (8x cut in L2 traffic), online softmax per wave. grid.y = b*9+h.
// ---------------------------------------------------------------------------
__global__ __launch_bounds__(256) void attn_kernel(
    const u16* __restrict__ Qr, const u16* __restrict__ Kr,
    const u16* __restrict__ Vt, u16* __restrict__ Ob)
{
    __shared__ __align__(16) u16 Klds[32 * KSTRIDE];   // K chunk: 32 kv x 64 d
    __shared__ __align__(16) u16 Vlds[64 * VSTRIDE];   // V chunk: 64 d x 32 kv
    __shared__ __align__(16) u16 Plds[8][16 * PSTRIDE];

    const int tid  = threadIdx.x;
    const int lane = tid & 31;
    const int wave = tid >> 5;
    const int l16  = lane & 15;
    const int half = lane >> 4;
    const int koff = half * 8;

    const int bh = blockIdx.y;
    const int b  = bh / NH;
    const int h  = bh % NH;
    const int kv = h / (NH / NKV);
    const int q0 = (blockIdx.x * 8 + wave) * 16;

    const u16* Qb = Qr + (size_t)bh * S_LEN * HD;
    const u16* Kb = Kr + (size_t)(b * NKV + kv) * S_LEN * HD;
    const u16* Vb = Vt + (size_t)(b * NKV + kv) * HD * S_LEN;

    const v16bf qa0 = load_a(Qb + (q0 + l16) * HD, koff);        // d 0..31
    const v16bf qa1 = load_a(Qb + (q0 + l16) * HD + 32, koff);   // d 32..63

    v8f acc[4] = {};
    float mrow[8], lrow[8];
#pragma unroll
    for (int r = 0; r < 8; ++r) { mrow[r] = -3.0e38f; lrow[r] = 0.0f; }

    u16* Pw = &Plds[wave][0];

    // cooperative-loader indices (256 threads)
    const int krow = tid >> 3, kpart = tid & 7;   // 32 rows x 8 uint4
    const int vrow = tid >> 2, vpart = tid & 3;   // 64 rows x 4 uint4

    for (int kv0 = 0; kv0 < S_LEN; kv0 += 32) {
        // stage K[kv0..kv0+31][0..63] and V^T[0..63][kv0..kv0+31] into LDS
        const u16* ksrc = Kb + (size_t)(kv0 + krow) * HD + kpart * 8;
        const u16* vsrc = Vb + (size_t)vrow * S_LEN + kv0 + vpart * 8;
        if (kv0 + 32 < S_LEN) {
            __builtin_prefetch(ksrc + 32 * HD, 0, 1);   // global_prefetch_b8
            __builtin_prefetch(vsrc + 32, 0, 1);
        }
        *(uint4*)(Klds + krow * KSTRIDE + kpart * 8) = *(const uint4*)ksrc;
        *(uint4*)(Vlds + vrow * VSTRIDE + vpart * 8) = *(const uint4*)vsrc;
        __syncthreads();

        // scores: two 16x16 tiles, K-dim = 64 (2 WMMA each), B from LDS
        v8f s0 = {}, s1 = {};
        {
            const u16* kr0 = Klds + l16 * KSTRIDE + half * 16;
            const u16* kr1 = Klds + (16 + l16) * KSTRIDE + half * 16;
            s0 = wmma_bf16(qa0, load_b(kr0), s0);
            s0 = wmma_bf16(qa1, load_b(kr0 + 32), s0);
            s1 = wmma_bf16(qa0, load_b(kr1), s1);
            s1 = wmma_bf16(qa1, load_b(kr1 + 32), s1);
        }

        // online softmax per row (rows r+8*half, spread over 16 lanes)
#pragma unroll
        for (int r = 0; r < 8; ++r) {
            float a0 = s0[r] * 0.125f, a1 = s1[r] * 0.125f;  // 1/sqrt(64)
            float mx = fmaxf(a0, a1);
            mx = fmaxf(mx, __shfl_xor(mx, 1));
            mx = fmaxf(mx, __shfl_xor(mx, 2));
            mx = fmaxf(mx, __shfl_xor(mx, 4));
            mx = fmaxf(mx, __shfl_xor(mx, 8));
            float mnew = fmaxf(mrow[r], mx);
            float p0 = __expf(a0 - mnew), p1 = __expf(a1 - mnew);
            float sum = p0 + p1;
            sum += __shfl_xor(sum, 1);
            sum += __shfl_xor(sum, 2);
            sum += __shfl_xor(sum, 4);
            sum += __shfl_xor(sum, 8);
            float scale = __expf(mrow[r] - mnew);
            lrow[r] = lrow[r] * scale + sum;
            mrow[r] = mnew;
#pragma unroll
            for (int c = 0; c < 4; ++c) acc[c][r] *= scale;
            int prow = r + 8 * half;
            Pw[prow * PSTRIDE + l16]      = f2bf(p0);
            Pw[prow * PSTRIDE + 16 + l16] = f2bf(p1);
        }
        // P write->read is intra-wave only (per-wave slice, DS ops in order)

        // P(16x32) @ V(32x64): P re-read in A layout, V from LDS
        v16bf pa = load_a(Pw + l16 * PSTRIDE, koff);
#pragma unroll
        for (int c = 0; c < 4; ++c)
            acc[c] = wmma_bf16(pa, load_b(Vlds + (c * 16 + l16) * VSTRIDE + half * 16),
                               acc[c]);
        __syncthreads();   // protect K/V chunk overwrite (uniform trip count)
    }

    // normalize and emit O in [b][s][h*64+d] (bf16 for the output projection)
#pragma unroll
    for (int r = 0; r < 8; ++r) {
        float inv = 1.0f / lrow[r];
        size_t row = (size_t)(b * S_LEN + q0 + r + 8 * half) * HID + h * HD;
#pragma unroll
        for (int c = 0; c < 4; ++c)
            Ob[row + c * 16 + l16] = f2bf(acc[c][r] * inv);
    }
}

// ---------------------------------------------------------------------------
// Kernel 3: output projection, fp32 result; weight chunk staged in LDS
// ---------------------------------------------------------------------------
__global__ __launch_bounds__(128) void oproj_kernel(
    const u16* __restrict__ Ob, const u16* __restrict__ Wob,
    float* __restrict__ out)
{
    __shared__ __align__(16) u16 Wl[64 * WSTRIDE];

    const int tid  = threadIdx.x;
    const int lane = tid & 31;
    const int wave = tid >> 5;
    const int l16  = lane & 15;
    const int half = lane >> 4;
    const int koff = half * 8;
    const int t0   = (blockIdx.x * 4 + wave) * 16;
    const int e0   = blockIdx.y * 64;

    v8f acc[4] = {};
    const u16* arow = Ob + (size_t)(t0 + l16) * HID;

    for (int kb = 0; kb < HID; kb += 32) {
#pragma unroll
        for (int i = tid; i < 256; i += 128) {
            int row = i >> 2, part = i & 3;
            const u16* src = Wob + (size_t)(e0 + row) * HID + kb + part * 8;
            if (kb + 32 < HID) __builtin_prefetch(src + 32, 0, 1);
            *(uint4*)(Wl + row * WSTRIDE + part * 8) = *(const uint4*)src;
        }
        __syncthreads();

        v16bf a = load_a(arow + kb, koff);
#pragma unroll
        for (int j = 0; j < 4; ++j)
            acc[j] = wmma_bf16(a, load_b(Wl + (j * 16 + l16) * WSTRIDE + half * 16),
                               acc[j]);
        __syncthreads();
    }
#pragma unroll
    for (int j = 0; j < 4; ++j)
#pragma unroll
        for (int r = 0; r < 8; ++r)
            out[(size_t)(t0 + r + 8 * half) * HID + e0 + j * 16 + l16] = acc[j][r];
}

// ---------------------------------------------------------------------------
extern "C" void kernel_launch(void* const* d_in, const int* in_sizes, int n_in,
                              void* d_out, int out_size, void* d_ws, size_t ws_size,
                              hipStream_t stream) {
    const float* x  = (const float*)d_in[0];
    const float* wq = (const float*)d_in[1];
    const float* wk = (const float*)d_in[2];
    const float* wv = (const float*)d_in[3];
    const float* wo = (const float*)d_in[4];
    float* out = (float*)d_out;

    const int nX  = TOK * HID;                 // 4718592
    const int nWq = HID * HID;                 // 331776
    const int nWk = (NKV * HD) * HID;          // 110592
    const int nWv = nWk;
    const int nWo = nWq;
    const int nQ  = NB * NH * S_LEN * HD;      // 4718592
    const int nK  = NB * NKV * S_LEN * HD;     // 1572864
    const int nV  = nK;

    u16* ws  = (u16*)d_ws;
    u16* Xb  = ws;            ws += nX;
    u16* Wqb = ws;            ws += nWq;
    u16* Wkb = ws;            ws += nWk;
    u16* Wvb = ws;            ws += nWv;
    u16* Wob = ws;            ws += nWo;
    u16* Qr  = ws;            ws += nQ;
    u16* Kr  = ws;            ws += nK;
    u16* Vt  = ws;            ws += nV;
    u16* Ob  = ws;            ws += nX;

    auto cvt = [&](const float* s, u16* d, int n) {
        cvt_f32_bf16<<<(n + 255) / 256, 256, 0, stream>>>(s, d, n);
    };
    cvt(x,  Xb,  nX);
    cvt(wq, Wqb, nWq);
    cvt(wk, Wkb, nWk);
    cvt(wv, Wvb, nWv);
    cvt(wo, Wob, nWo);

    qkv_rope_kernel<<<dim3(TOK / 64, NH + 2 * NKV), 128, 0, stream>>>(
        Xb, Wqb, Wkb, Wvb, Qr, Kr, Vt);

    attn_kernel<<<dim3(S_LEN / 128, NB * NH), 256, 0, stream>>>(Qr, Kr, Vt, Ob);

    oproj_kernel<<<dim3(TOK / 64, HID / 64), 128, 0, stream>>>(Ob, Wob, out);
}
